// TwelveSitesNN2_3590592659557
// MI455X (gfx1250) — matrix operand order repaired and checked by reading, not compiled, not run on hardware
//
#include <hip/hip_runtime.h>

typedef __attribute__((ext_vector_type(16))) _Float16 v16h;
typedef __attribute__((ext_vector_type(8)))  _Float16 v8h;
typedef __attribute__((ext_vector_type(8)))  float    v8f;

#define NLAYER 63
#define NS 12
#define TB 16                 // batch samples per workgroup
#define ROWS (TB*NS)          // 192 GEMM rows for the CNN stages
#define NTHREADS 256          // 8 wave32

// ---- LDS byte offsets (all 32B aligned) ----
#define OFF_WB1   0           // [32][32] f16, w1 col-major (n,k), k>=6 zero   (2048 B)
#define OFF_WB2   2048        // [64][96] f16, w2 col-major                     (12288 B)
#define OFF_FW1   14336       // [64][128] f16, fw1 col-major                   (16384 B)
#define OFF_FW2   30720       // [64][64] f16, fw2 col-major                    (8192 B)
#define OFF_FW3   38912       // [64] f16                                       (128 B)
#define OFF_FB1   39040       // [64] f32                                       (256 B)
#define OFF_FB2   39296       // [64] f32                                       (256 B)
#define OFF_FB3   39552       // [1] f32 (padded)                               (32 B)
#define OFF_ALLC2 39584       // [192][96] f16                                  (36864 B)
#define OFF_R2    76448       // reused region
#define OFF_XS    OFF_R2            // [16][12][2] f32                          (1536 B)
#define OFF_ALLC1 (OFF_R2+1536)     // [192][8] f16                             (3072 B)
#define OFF_H2    OFF_R2            // [192][64] f16 (aliases xs/allc1)         (24576 B)
#define OFF_POOL  (OFF_R2+24576)    // [16][128] f16                            (4096 B)
#define OFF_A1    (OFF_R2+28672)    // [16][64] f16                             (2048 B)
#define OFF_A2    (OFF_R2+30720)    // [16][64] f16                             (2048 B)
#define SMEM_BYTES 109216

// A fragment for v_wmma_f32_16x16x32_f16 from row-major LDS [.][stride] f16.
// lane row = row0 + (lane&15); elems 0..7 -> K = kbase + (lane>=16?8:0) + i,
// elems 8..15 -> +16 (per CDNA5 ISA 16-bit A 16x32 layout).
__device__ __forceinline__ v16h lds_afrag(const _Float16* m, int row0, int stride,
                                          int kbase, int lane) {
  const _Float16* r = m + (size_t)(row0 + (lane & 15)) * stride + kbase + ((lane >> 4) << 3);
  v8h lo = *(const v8h*)r;
  v8h hi = *(const v8h*)(r + 16);
  v16h a;
#pragma unroll
  for (int i = 0; i < 8; ++i) { a[i] = lo[i]; a[i + 8] = hi[i]; }
  return a;
}

// B fragment from column-major LDS weights [N][kstride] f16.
// lane col = col0 + (lane&15); elems 0..7 -> K = kbase + (lane>=16?8:0) + i, elems 8..15 -> +16.
__device__ __forceinline__ v16h lds_bfrag(const _Float16* w, int col0, int kstride,
                                          int kbase, int lane) {
  const _Float16* c = w + (size_t)(col0 + (lane & 15)) * kstride + kbase + ((lane >> 4) << 3);
  v8h lo = *(const v8h*)c;
  v8h hi = *(const v8h*)(c + 16);
  v16h b;
#pragma unroll
  for (int i = 0; i < 8; ++i) { b[i] = lo[i]; b[i + 8] = hi[i]; }
  return b;
}

__global__ __launch_bounds__(NTHREADS)
void twelve_sites_nn(const float* __restrict__ zs,
                     const float* __restrict__ w1g,
                     const float* __restrict__ w2g,
                     const float* __restrict__ fw1g, const float* __restrict__ fb1g,
                     const float* __restrict__ fw2g, const float* __restrict__ fb2g,
                     const float* __restrict__ fw3g, const float* __restrict__ fb3g,
                     float* __restrict__ out)
{
  extern __shared__ char smem[];
  _Float16* wB1   = (_Float16*)(smem + OFF_WB1);
  _Float16* wB2   = (_Float16*)(smem + OFF_WB2);
  _Float16* fw1c  = (_Float16*)(smem + OFF_FW1);
  _Float16* fw2c  = (_Float16*)(smem + OFF_FW2);
  _Float16* fw3h  = (_Float16*)(smem + OFF_FW3);
  float*    fb1f  = (float*)(smem + OFF_FB1);
  float*    fb2f  = (float*)(smem + OFF_FB2);
  float*    fb3f  = (float*)(smem + OFF_FB3);
  float*    xs    = (float*)(smem + OFF_XS);
  _Float16* allc1 = (_Float16*)(smem + OFF_ALLC1);
  _Float16* allc2 = (_Float16*)(smem + OFF_ALLC2);
  _Float16* h2    = (_Float16*)(smem + OFF_H2);
  _Float16* pld   = (_Float16*)(smem + OFF_POOL);
  _Float16* a1    = (_Float16*)(smem + OFF_A1);
  _Float16* a2    = (_Float16*)(smem + OFF_A2);

  const int tid  = threadIdx.x;
  const int lane = tid & 31;
  const int wave = tid >> 5;
  const int l    = blockIdx.y;
  const int b0   = blockIdx.x * TB;

  // ---------- Phase 0: stage layer-l weights into LDS (col-major, f16) ----------
  for (int i = tid; i < 32 * 32; i += NTHREADS) {        // w1 (63,6,32) -> [n][k] pad K 6->32
    int n = i >> 5, k = i & 31;
    float v = (k < 6) ? w1g[(size_t)l * 192 + k * 32 + n] : 0.0f;
    wB1[n * 32 + k] = (_Float16)v;
  }
  for (int i = tid; i < 64 * 96; i += NTHREADS) {        // w2 (63,96,64) -> [n][k]
    int n = i / 96, k = i - n * 96;
    wB2[n * 96 + k] = (_Float16)w2g[(size_t)l * 6144 + k * 64 + n];
  }
  for (int i = tid; i < 64 * 128; i += NTHREADS) {       // fw1 (63,128,64) -> [n][k]
    int n = i >> 7, k = i & 127;
    fw1c[n * 128 + k] = (_Float16)fw1g[(size_t)l * 8192 + k * 64 + n];
  }
  for (int i = tid; i < 64 * 64; i += NTHREADS) {        // fw2 (63,64,64) -> [n][k]
    int n = i >> 6, k = i & 63;
    fw2c[n * 64 + k] = (_Float16)fw2g[(size_t)l * 4096 + k * 64 + n];
  }
  if (tid < 64) {
    fw3h[tid] = (_Float16)fw3g[(size_t)l * 64 + tid];
    fb1f[tid] = fb1g[(size_t)l * 64 + tid];
    fb2f[tid] = fb2g[(size_t)l * 64 + tid];
  }
  if (tid == 0) fb3f[0] = fb3g[l];

  // ---------- Phase 0b: stage zs tile, build allc1 (ring neighbor sums, K=6) ----------
  for (int i = tid; i < TB * NS * 2; i += NTHREADS) {    // zs (B,63,12,2)
    int b = i / 24; int rem = i - b * 24; int s = rem >> 1; int c = rem & 1;
    xs[i] = zs[(((size_t)(b0 + b) * NLAYER + l) * NS + s) * 2 + c];
  }
  __syncthreads();
  if (tid < ROWS) {
    int b = tid / NS, s = tid - b * NS;
    const float* xb = xs + b * 24;
#pragma unroll
    for (int c = 0; c < 2; ++c) {
      float x0 = xb[s * 2 + c];
      float sB = xb[((s + 1) % NS) * 2 + c] + xb[((s + 3) % NS) * 2 + c]
               + xb[((s + 9) % NS) * 2 + c] + xb[((s + 11) % NS) * 2 + c];
      float sC = xb[((s + 2) % NS) * 2 + c] + xb[((s + 4) % NS) * 2 + c]
               + xb[((s + 8) % NS) * 2 + c] + xb[((s + 10) % NS) * 2 + c];
      allc1[tid * 8 + 0 + c] = (_Float16)x0;
      allc1[tid * 8 + 2 + c] = (_Float16)sB;
      allc1[tid * 8 + 4 + c] = (_Float16)sC;
    }
    allc1[tid * 8 + 6] = (_Float16)0.0f;
    allc1[tid * 8 + 7] = (_Float16)0.0f;
  }
  __syncthreads();

  // ---------- GEMM1 (192x32x6->32, WMMA, zero-padded K) -> relu -> allc2[:, :32] ----------
  for (int t = wave; t < 24; t += 8) {
    int mt = t >> 1, nt = t & 1;
    v16h a;
#pragma unroll
    for (int i = 0; i < 16; ++i) a[i] = (_Float16)0.0f;
    if (lane < 16) {                                     // lanes>=16 hold K=8..15 => zero
      v8h lo = *(const v8h*)(allc1 + (mt * 16 + lane) * 8);
#pragma unroll
      for (int i = 0; i < 8; ++i) a[i] = lo[i];
    }
    v16h b = lds_bfrag(wB1, nt * 16, 32, 0, lane);
    v8f acc;
#pragma unroll
    for (int i = 0; i < 8; ++i) acc[i] = 0.0f;
    acc = __builtin_amdgcn_wmma_f32_16x16x32_f16(false, a, false, b, (short)0, acc, false, false);
    int rw  = mt * 16 + ((lane >> 4) << 3);
    int col = nt * 16 + (lane & 15);
#pragma unroll
    for (int r = 0; r < 8; ++r)
      allc2[(rw + r) * 96 + col] = (_Float16)fmaxf(acc[r], 0.0f);
  }
  __syncthreads();

  // ---------- expand allc2[:, 32:96] from allc2[:, :32] (ring sums on h1) ----------
  for (int i = tid; i < ROWS * 32; i += NTHREADS) {
    int row = i >> 5, c = i & 31;
    int b = row / NS, s = row - b * NS;
    const _Float16* hb = allc2 + (size_t)(b * NS) * 96;
    float v1 = (float)hb[((s + 1) % NS) * 96 + c] + (float)hb[((s + 3) % NS) * 96 + c]
             + (float)hb[((s + 9) % NS) * 96 + c] + (float)hb[((s + 11) % NS) * 96 + c];
    float v2 = (float)hb[((s + 2) % NS) * 96 + c] + (float)hb[((s + 4) % NS) * 96 + c]
             + (float)hb[((s + 8) % NS) * 96 + c] + (float)hb[((s + 10) % NS) * 96 + c];
    allc2[row * 96 + 32 + c] = (_Float16)v1;
    allc2[row * 96 + 64 + c] = (_Float16)v2;
  }
  __syncthreads();

  // ---------- GEMM2 (192x64x96, WMMA, 3 K-steps) -> relu -> h2 ----------
  for (int t = wave; t < 48; t += 8) {
    int mt = t >> 2, nt = t & 3;
    v8f acc;
#pragma unroll
    for (int i = 0; i < 8; ++i) acc[i] = 0.0f;
#pragma unroll
    for (int ks = 0; ks < 3; ++ks) {
      v16h a = lds_afrag(allc2, mt * 16, 96, ks * 32, lane);
      v16h b = lds_bfrag(wB2, nt * 16, 96, ks * 32, lane);
      acc = __builtin_amdgcn_wmma_f32_16x16x32_f16(false, a, false, b, (short)0, acc, false, false);
    }
    int rw  = mt * 16 + ((lane >> 4) << 3);
    int col = nt * 16 + (lane & 15);
#pragma unroll
    for (int r = 0; r < 8; ++r)
      h2[(rw + r) * 64 + col] = (_Float16)fmaxf(acc[r], 0.0f);
  }
  __syncthreads();

  // ---------- pool: mean + max over 12 sites -> pld [16][128] ----------
  for (int i = tid; i < TB * 64; i += NTHREADS) {
    int b = i >> 6, c = i & 63;
    float sum = 0.0f, mx = 0.0f;                         // relu output >= 0
#pragma unroll
    for (int s = 0; s < NS; ++s) {
      float v = (float)h2[(b * NS + s) * 64 + c];
      sum += v; mx = fmaxf(mx, v);
    }
    pld[b * 128 + c]      = (_Float16)(sum * (1.0f / 12.0f));
    pld[b * 128 + 64 + c] = (_Float16)mx;
  }
  __syncthreads();

  // ---------- FC1 (16x64x128, WMMA, 4 K-steps, waves 0..3) + bias + relu ----------
  if (wave < 4) {
    int nt = wave;
    v8f acc;
#pragma unroll
    for (int i = 0; i < 8; ++i) acc[i] = 0.0f;
#pragma unroll
    for (int ks = 0; ks < 4; ++ks) {
      v16h a = lds_afrag(pld, 0, 128, ks * 32, lane);
      v16h b = lds_bfrag(fw1c, nt * 16, 128, ks * 32, lane);
      acc = __builtin_amdgcn_wmma_f32_16x16x32_f16(false, a, false, b, (short)0, acc, false, false);
    }
    int rw  = (lane >> 4) << 3;
    int col = nt * 16 + (lane & 15);
#pragma unroll
    for (int r = 0; r < 8; ++r)
      a1[(rw + r) * 64 + col] = (_Float16)fmaxf(acc[r] + fb1f[col], 0.0f);
  }
  __syncthreads();

  // ---------- FC2 (16x64x64, WMMA, 2 K-steps) + bias + relu ----------
  if (wave < 4) {
    int nt = wave;
    v8f acc;
#pragma unroll
    for (int i = 0; i < 8; ++i) acc[i] = 0.0f;
#pragma unroll
    for (int ks = 0; ks < 2; ++ks) {
      v16h a = lds_afrag(a1, 0, 64, ks * 32, lane);
      v16h b = lds_bfrag(fw2c, nt * 16, 64, ks * 32, lane);
      acc = __builtin_amdgcn_wmma_f32_16x16x32_f16(false, a, false, b, (short)0, acc, false, false);
    }
    int rw  = (lane >> 4) << 3;
    int col = nt * 16 + (lane & 15);
#pragma unroll
    for (int r = 0; r < 8; ++r)
      a2[(rw + r) * 64 + col] = (_Float16)fmaxf(acc[r] + fb2f[col], 0.0f);
  }
  __syncthreads();

  // ---------- FC3: 64-dot per sample -> out (B, 63) ----------
  if (tid < TB) {
    float s = fb3f[0];
    const _Float16* row = a2 + tid * 64;
#pragma unroll
    for (int k = 0; k < 64; ++k) s += (float)row[k] * (float)fw3h[k];
    out[(size_t)(b0 + tid) * NLAYER + l] = s;
  }
}

extern "C" void kernel_launch(void* const* d_in, const int* in_sizes, int n_in,
                              void* d_out, int out_size, void* d_ws, size_t ws_size,
                              hipStream_t stream) {
  const float* zs  = (const float*)d_in[0];
  const float* w1  = (const float*)d_in[1];
  const float* w2  = (const float*)d_in[2];
  const float* fw1 = (const float*)d_in[3];
  const float* fb1 = (const float*)d_in[4];
  const float* fw2 = (const float*)d_in[5];
  const float* fb2 = (const float*)d_in[6];
  const float* fw3 = (const float*)d_in[7];
  const float* fb3 = (const float*)d_in[8];
  float* out = (float*)d_out;

  int Bn = in_sizes[0] / (NLAYER * NS * 2);   // 2048
  dim3 grid(Bn / TB, NLAYER);                  // (128, 63)
  twelve_sites_nn<<<grid, NTHREADS, SMEM_BYTES, stream>>>(
      zs, w1, w2, fw1, fb1, fw2, fb2, fw3, fb3, out);
  (void)n_in; (void)out_size; (void)d_ws; (void)ws_size;
}
